// CrfRnnLayer_20933670600704
// MI455X (gfx1250) — matrix operand order repaired and checked by reading, not compiled, not run on hardware
//
#include <hip/hip_runtime.h>

#define B_  4
#define H_  64
#define W_  64
#define C_  21
#define N_  4096
#define CP  32   // channels padded to 32 for two 16-row WMMA tiles

typedef __attribute__((ext_vector_type(16))) _Float16 v16h;
typedef __attribute__((ext_vector_type(8)))  _Float16 v8h;
typedef __attribute__((ext_vector_type(8)))  float    v8f;

// ---- WMMA fragment loaders (CDNA5 16-bit layouts, wave32) -------------------
// A (16x32 MxK): lane m = lane&15, sel = lane>>4.
//   elements 0..7  hold K = sel*8 + i        (VGPR0-3)
//   elements 8..15 hold K = 16 + sel*8 + i   (VGPR4-7)
__device__ __forceinline__ v16h load_fragA(const _Float16* row, int sel) {
  v8h lo = *(const v8h*)(row + sel * 8);
  v8h hi = *(const v8h*)(row + 16 + sel * 8);
  v16h a;
#pragma unroll
  for (int i = 0; i < 8; ++i) { a[i] = lo[i]; a[i + 8] = hi[i]; }
  return a;
}
// B (32x16 KxN): lane n = lane&15; elements 0..7 hold K = sel*8 + i, elements 8..15 hold K = 16 + sel*8 + i.
// We exploit K symmetry: B[k][n] = K[n][k], so each lane reads row n contiguously.
__device__ __forceinline__ v16h load_fragB(const _Float16* row, int sel) {
  v8h lo = *(const v8h*)(row + sel * 8);
  v8h hi = *(const v8h*)(row + 16 + sel * 8);
  v16h bfr;
#pragma unroll
  for (int i = 0; i < 8; ++i) { bfr[i] = lo[i]; bfr[i + 8] = hi[i]; }
  return bfr;
}

// ---- Kernel builders --------------------------------------------------------
__global__ void build_spatial(_Float16* __restrict__ Ks, float* __restrict__ s_norm) {
  const int n = blockIdx.x;
  const int t = threadIdx.x;
  const float yn = (float)(n >> 6), xn = (float)(n & 63);
  const float inv_g2 = 1.0f / 9.0f;  // 1/theta_gamma^2
  float sum = 0.0f;
  for (int m = t; m < N_; m += 256) {
    float dy = yn - (float)(m >> 6);
    float dx = xn - (float)(m & 63);
    float v = __expf(-0.5f * (dy * dy + dx * dx) * inv_g2);
    _Float16 h = (_Float16)v;
    Ks[(size_t)n * N_ + m] = h;
    sum += (float)h;
  }
  __shared__ float red[256];
  red[t] = sum; __syncthreads();
  for (int s = 128; s > 0; s >>= 1) { if (t < s) red[t] += red[t + s]; __syncthreads(); }
  if (t == 0) s_norm[n] = red[0];
}

__global__ void build_bilateral(const float* __restrict__ rgb,
                                _Float16* __restrict__ Kb, float* __restrict__ b_norm) {
  const int row = blockIdx.x;           // b*N + n
  const int b = row >> 12, n = row & 4095;
  const int t = threadIdx.x;
  const float inv_a = 1.0f / 160.0f, inv_bt = 1.0f / 3.0f;
  const float* rb = rgb + (size_t)b * N_ * 3;
  const float f0 = (float)(n >> 6) * inv_a;
  const float f1 = (float)(n & 63) * inv_a;
  const float f2 = rb[n * 3 + 0] * inv_bt;
  const float f3 = rb[n * 3 + 1] * inv_bt;
  const float f4 = rb[n * 3 + 2] * inv_bt;
  _Float16* krow = Kb + (size_t)row * N_;
  float sum = 0.0f;
  for (int m = t; m < N_; m += 256) {
    float d0 = f0 - (float)(m >> 6) * inv_a;
    float d1 = f1 - (float)(m & 63) * inv_a;
    float d2 = f2 - rb[m * 3 + 0] * inv_bt;
    float d3 = f3 - rb[m * 3 + 1] * inv_bt;
    float d4 = f4 - rb[m * 3 + 2] * inv_bt;
    float v = __expf(-0.5f * (d0*d0 + d1*d1 + d2*d2 + d3*d3 + d4*d4));
    _Float16 h = (_Float16)v;
    krow[m] = h;
    sum += (float)h;
  }
  __shared__ float red[256];
  red[t] = sum; __syncthreads();
  for (int s = 128; s > 0; s >>= 1) { if (t < s) red[t] += red[t + s]; __syncthreads(); }
  if (t == 0) b_norm[row] = red[0];
}

// ---- Precompute compat@Ws and compat@Wb, padded to 32x32 --------------------
__global__ void mix_mats(const float* __restrict__ Ws, const float* __restrict__ Wb,
                         const float* __restrict__ compat,
                         float* __restrict__ CWs, float* __restrict__ CWb) {
  int c = threadIdx.x >> 5;
  int k = threadIdx.x & 31;
  float a = 0.0f, bb = 0.0f;
  if (c < C_ && k < C_) {
    for (int j = 0; j < C_; ++j) {
      a  += compat[c * C_ + j] * Ws[j * C_ + k];
      bb += compat[c * C_ + j] * Wb[j * C_ + k];
    }
  }
  CWs[c * 32 + k] = a;
  CWb[c * 32 + k] = bb;
}

// ---- p0 = softmax(u); zero pad rows of both P buffers -----------------------
__global__ void init_p(const float* __restrict__ unary,
                       _Float16* __restrict__ P0, _Float16* __restrict__ P1) {
  int p = blockIdx.x * blockDim.x + threadIdx.x;
  if (p >= B_ * N_) return;
  int b = p >> 12, n = p & 4095;
  const float* u = unary + (size_t)p * C_;
  float mx = -3.402823466e38f;
  for (int c = 0; c < C_; ++c) mx = fmaxf(mx, u[c]);
  float e[C_], s = 0.0f;
  for (int c = 0; c < C_; ++c) { e[c] = __expf(u[c] - mx); s += e[c]; }
  float inv = 1.0f / s;
  _Float16* b0 = P0 + (size_t)b * CP * N_ + n;
  _Float16* b1 = P1 + (size_t)b * CP * N_ + n;
  for (int c = 0; c < C_; ++c) b0[(size_t)c * N_] = (_Float16)(e[c] * inv);
  for (int c = C_; c < CP; ++c) {
    b0[(size_t)c * N_] = (_Float16)0.0f;
    b1[(size_t)c * N_] = (_Float16)0.0f;
  }
}

// ---- One fused mean-field iteration -----------------------------------------
// One wave owns 16 pixels (n0..n0+15) of one batch; computes all 32 padded
// channels of sp and bl via WMMA, normalizes, applies compat@W mixing, then
// softmax->Pout (or q->qout on last iteration).
__global__ void __launch_bounds__(256)
crf_iter(const float* __restrict__ unary,
         const _Float16* __restrict__ Ks, const _Float16* __restrict__ Kb,
         const float* __restrict__ s_norm, const float* __restrict__ b_norm,
         const float* __restrict__ CWs, const float* __restrict__ CWb,
         const _Float16* __restrict__ Pin, _Float16* __restrict__ Pout,
         float* __restrict__ qout, int last) {
  __shared__ float SPl[8][CP][17];
  __shared__ float BLl[8][CP][17];
  __shared__ float Ql[8][C_][17];

  const int tid  = threadIdx.x;
  const int wv   = tid >> 5;
  const int lane = tid & 31;
  const int sel  = lane >> 4;
  const int nl   = lane & 15;

  const int gw = blockIdx.x * 8 + wv;   // 0..1023
  const int b  = gw >> 8;
  const int n0 = (gw & 255) << 4;

  const _Float16* Pb  = Pin + (size_t)b * CP * N_;
  const _Float16* pa0 = Pb + (size_t)nl * N_;          // channel rows 0-15
  const _Float16* pa1 = Pb + (size_t)(nl + 16) * N_;   // channel rows 16-31
  const _Float16* pks = Ks + (size_t)(n0 + nl) * N_;   // symmetric: row n == col n
  const _Float16* pkb = Kb + ((size_t)b * N_ + n0 + nl) * N_;

  v8f sp0 = {}, sp1 = {}, bl0 = {}, bl1 = {};
#pragma unroll 2
  for (int k0 = 0; k0 < N_; k0 += 32) {
    v16h a0 = load_fragA(pa0 + k0, sel);
    v16h a1 = load_fragA(pa1 + k0, sel);
    v16h bs = load_fragB(pks + k0, sel);
    v16h bb = load_fragB(pkb + k0, sel);
    sp0 = __builtin_amdgcn_wmma_f32_16x16x32_f16(false, a0, false, bs, (short)0, sp0, false, false);
    sp1 = __builtin_amdgcn_wmma_f32_16x16x32_f16(false, a1, false, bs, (short)0, sp1, false, false);
    bl0 = __builtin_amdgcn_wmma_f32_16x16x32_f16(false, a0, false, bb, (short)0, bl0, false, false);
    bl1 = __builtin_amdgcn_wmma_f32_16x16x32_f16(false, a1, false, bb, (short)0, bl1, false, false);
  }

  // normalize per pixel column; D layout: element r -> (m = r + 8*sel, n = nl)
  const int nglob = n0 + nl;
  const float sdiv = 1.0f / s_norm[nglob];
  const float bdiv = 1.0f / b_norm[b * N_ + nglob];
#pragma unroll
  for (int r = 0; r < 8; ++r) {
    SPl[wv][r + 8 * sel][nl]      = sp0[r] * sdiv;
    SPl[wv][16 + r + 8 * sel][nl] = sp1[r] * sdiv;
    BLl[wv][r + 8 * sel][nl]      = bl0[r] * bdiv;
    BLl[wv][16 + r + 8 * sel][nl] = bl1[r] * bdiv;
  }
  __syncthreads();

  // q[c,n] = u[c,n] - sum_k ( (compat@Ws)[c,k]*sp[k,n] + (compat@Wb)[c,k]*bl[k,n] )
  const float* uptr = unary + ((size_t)b * N_ + nglob) * C_;
  const int cbase = sel * 16;
#pragma unroll
  for (int ci = 0; ci < 16; ++ci) {
    int c = cbase + ci;
    if (c < C_) {
      float acc = 0.0f;
#pragma unroll
      for (int k = 0; k < C_; ++k)
        acc += CWs[c * 32 + k] * SPl[wv][k][nl] + CWb[c * 32 + k] * BLl[wv][k][nl];
      Ql[wv][c][nl] = uptr[c] - acc;
    }
  }
  __syncthreads();

  if (lane < 16) {
    const int n = n0 + lane;
    if (last) {
      float* o = qout + ((size_t)b * N_ + n) * C_;
      for (int c = 0; c < C_; ++c) o[c] = Ql[wv][c][lane];
    } else {
      float mx = -3.402823466e38f;
      for (int c = 0; c < C_; ++c) mx = fmaxf(mx, Ql[wv][c][lane]);
      float e[C_], s = 0.0f;
      for (int c = 0; c < C_; ++c) { e[c] = __expf(Ql[wv][c][lane] - mx); s += e[c]; }
      float inv = 1.0f / s;
      _Float16* po = Pout + (size_t)b * CP * N_ + n;
      for (int c = 0; c < C_; ++c) po[(size_t)c * N_] = (_Float16)(e[c] * inv);
    }
  }
}

// ---- Host-side orchestration ------------------------------------------------
extern "C" void kernel_launch(void* const* d_in, const int* in_sizes, int n_in,
                              void* d_out, int out_size, void* d_ws, size_t ws_size,
                              hipStream_t stream) {
  (void)in_sizes; (void)n_in; (void)out_size; (void)ws_size;
  const float* unary  = (const float*)d_in[0];
  const float* rgb    = (const float*)d_in[1];
  const float* Ws     = (const float*)d_in[2];
  const float* Wb     = (const float*)d_in[3];
  const float* compat = (const float*)d_in[4];
  float* out = (float*)d_out;

  char* ws = (char*)d_ws;
  size_t off = 0;
  _Float16* Ks = (_Float16*)(ws + off); off += (size_t)N_ * N_ * 2;           // 32 MB
  _Float16* Kb = (_Float16*)(ws + off); off += (size_t)B_ * N_ * N_ * 2;      // 128 MB
  _Float16* P0 = (_Float16*)(ws + off); off += (size_t)B_ * CP * N_ * 2;      // 1 MB
  _Float16* P1 = (_Float16*)(ws + off); off += (size_t)B_ * CP * N_ * 2;      // 1 MB
  float* s_norm = (float*)(ws + off); off += (size_t)N_ * 4;
  float* b_norm = (float*)(ws + off); off += (size_t)B_ * N_ * 4;
  float* CWs    = (float*)(ws + off); off += 32 * 32 * 4;
  float* CWb    = (float*)(ws + off); off += 32 * 32 * 4;

  build_spatial<<<N_, 256, 0, stream>>>(Ks, s_norm);
  build_bilateral<<<B_ * N_, 256, 0, stream>>>(rgb, Kb, b_norm);
  mix_mats<<<1, 1024, 0, stream>>>(Ws, Wb, compat, CWs, CWb);
  init_p<<<(B_ * N_) / 128, 128, 0, stream>>>(unary, P0, P1);

  // 5 mean-field iterations, double-buffered P
  crf_iter<<<128, 256, 0, stream>>>(unary, Ks, Kb, s_norm, b_norm, CWs, CWb, P0, P1, out, 0);
  crf_iter<<<128, 256, 0, stream>>>(unary, Ks, Kb, s_norm, b_norm, CWs, CWb, P1, P0, out, 0);
  crf_iter<<<128, 256, 0, stream>>>(unary, Ks, Kb, s_norm, b_norm, CWs, CWb, P0, P1, out, 0);
  crf_iter<<<128, 256, 0, stream>>>(unary, Ks, Kb, s_norm, b_norm, CWs, CWb, P1, P0, out, 0);
  crf_iter<<<128, 256, 0, stream>>>(unary, Ks, Kb, s_norm, b_norm, CWs, CWb, P0, P1, out, 1);
}
